// FullDeformF2F_81879256531317
// MI455X (gfx1250) — matrix operand order repaired and checked by reading, not compiled, not run on hardware
//
#include <hip/hip_runtime.h>
#include <hip/hip_bf16.h>

typedef __attribute__((ext_vector_type(16))) _Float16 v16h;
typedef __attribute__((ext_vector_type(8)))  _Float16 v8h;
typedef __attribute__((ext_vector_type(8)))  float    v8f;

// ---------------------------------------------------------------------------
// Pack f32 weights (Co, C, kh, kw) into f16 swizzled for the WMMA B-fragment:
// flat index = ((ntile*KT + t)*32 + lane)*16 + j
//   maps to  oc = ntile*16 + (lane&15),  k = t*32 + (lane>>4)*16 + j
// where k = c*K2 + k2 (row-major over (C,kh,kw)), K = C*K2, KT = K/32.
// ---------------------------------------------------------------------------
__global__ void pack_weights_kernel(const float* __restrict__ wd,
                                    _Float16* __restrict__ wpack,
                                    int Co, int K) {
    int KT = K >> 5;
    size_t idx = (size_t)blockIdx.x * blockDim.x + threadIdx.x;
    size_t total = (size_t)Co * K;
    if (idx >= total) return;
    int j = (int)(idx & 15);
    size_t q = idx >> 4;
    int lanei = (int)(q & 31); q >>= 5;
    int t = (int)(q % KT);
    int ntile = (int)(q / KT);
    int k  = t * 32 + ((lanei >> 4) << 4) + j;
    int oc = (ntile << 4) + (lanei & 15);
    wpack[idx] = (_Float16)wd[(size_t)oc * K + k];
}

// ---------------------------------------------------------------------------
// Offset conv: regular conv2d, Cout2 = 2*K2 output channels (<=18). Plain f32.
// ---------------------------------------------------------------------------
template <int KS>
__global__ void offset_conv_kernel(const float* __restrict__ x,
                                   const float* __restrict__ wo,
                                   const float* __restrict__ bo,
                                   float* __restrict__ out,
                                   int B, int Cin, int Cout2, int H, int W, int pad) {
    size_t idx = (size_t)blockIdx.x * blockDim.x + threadIdx.x;
    size_t total = (size_t)B * Cout2 * H * W;
    if (idx >= total) return;
    int wcol = (int)(idx % W); size_t r = idx / W;
    int h  = (int)(r % H); r /= H;
    int oc = (int)(r % Cout2);
    int b  = (int)(r / Cout2);
    int HW = H * W;
    float acc = bo[oc];
    const float* wrow = wo + (size_t)oc * Cin * KS * KS;
    const float* xb   = x + (size_t)b * Cin * HW;
    for (int c = 0; c < Cin; ++c) {
        const float* xp = xb + (size_t)c * HW;
        const float* wp = wrow + c * KS * KS;
#pragma unroll
        for (int ky = 0; ky < KS; ++ky) {
            int yy = h + ky - pad;
            if (yy < 0 || yy >= H) continue;
#pragma unroll
            for (int kx = 0; kx < KS; ++kx) {
                int xx = wcol + kx - pad;
                if (xx < 0 || xx >= W) continue;
                acc += wp[ky * KS + kx] * xp[yy * W + xx];
            }
        }
    }
    out[idx] = acc;
}

// ---------------------------------------------------------------------------
// Deformable conv as WMMA GEMM.
//   M = B*H*W (64 rows per workgroup), N = Co (NT 16-col tiles), K = C*K2.
// Workgroup: 128 threads = 4 waves. Wave w owns M-rows [w*16, w*16+16) of the
// 64-row tile and computes ALL NT N-tiles (acc lives in VGPRs).
// Per K-chunk of 32:
//   - B chunk (NT KB) staged LDS-wide ONCE via global_load_async_to_lds_b128
//     (ASYNCcnt), overlapped with the bilinear gather of the 64x32 A-tile.
//   - s_wait_asynccnt 0 + barrier, then each wave: 1 A-frag (ds_load_b128 x2)
//     reused across NT WMMAs, B-frags straight from LDS.
// Each bilinear sample is gathered exactly once; B is read from L2 once per
// 64 output rows (4x less than a 16-row tile).
// ---------------------------------------------------------------------------
template <int K2, int KS, int NT>
__global__ __launch_bounds__(128) void deform_wmma_kernel(
    const float* __restrict__ x, const float* __restrict__ off,
    const _Float16* __restrict__ wpack, const float* __restrict__ bias,
    float* __restrict__ out, int B, int C, int Co, int H, int W,
    int pad, int relu) {
    const int HW = H * W;
    const int KT = (C * K2) >> 5;     // K / 32
    const int tid  = threadIdx.x;
    const int lane = tid & 31;
    const int wv   = tid >> 5;

    __shared__ float sWgt[4][64 * K2];
    __shared__ int   sIdx[4][64 * K2];
    __shared__ int   sB[64];
    __shared__ __attribute__((aligned(32))) _Float16 Alds[64 * 32];
    __shared__ __attribute__((aligned(32))) _Float16 Bs[NT * 512];

    const unsigned bsBase = (unsigned)(size_t)(void*)&Bs[0];

    // ---- per-tile bilinear corner precompute (64 positions x K2 taps) ----
    for (int ci = tid; ci < 64 * K2; ci += 128) {
        int m  = ci & 63;
        int k2 = ci >> 6;
        int pos = blockIdx.x * 64 + m;
        int b   = pos / HW;
        int rem = pos - b * HW;
        int h   = rem / W;
        int wc  = rem - h * W;
        if (k2 == 0) sB[m] = b;
        int ky = k2 / KS, kx = k2 % KS;
        float o0 = off[((size_t)b * (2 * K2) + 2 * k2) * HW + rem];
        float o1 = off[((size_t)b * (2 * K2) + 2 * k2 + 1) * HW + rem];
        float py = (float)(h + ky - pad) + o0;
        float px = (float)(wc + kx - pad) + o1;
        float y0f = floorf(py), x0f = floorf(px);
        float wy = py - y0f, wx = px - x0f;
        int y0 = (int)y0f, x0 = (int)x0f;
#pragma unroll
        for (int dy = 0; dy < 2; ++dy) {
#pragma unroll
            for (int dx = 0; dx < 2; ++dx) {
                int yy = y0 + dy, xx = x0 + dx;
                bool valid = (yy >= 0) && (yy < H) && (xx >= 0) && (xx < W);
                float wgt = (dy ? wy : 1.0f - wy) * (dx ? wx : 1.0f - wx);
                int yc = min(max(yy, 0), H - 1);
                int xc = min(max(xx, 0), W - 1);
                sWgt[dy * 2 + dx][ci] = valid ? wgt : 0.0f;
                sIdx[dy * 2 + dx][ci] = yc * W + xc;
            }
        }
    }
    __syncthreads();

    v8f acc[NT];
    const v8f vzero = {0.f, 0.f, 0.f, 0.f, 0.f, 0.f, 0.f, 0.f};
#pragma unroll
    for (int i = 0; i < NT; ++i) acc[i] = vzero;

    const int off8 = (lane >> 4) * 16;  // contiguous A-fragment K offset (k = 16h + s)
    const int mA   = lane & 15;

    for (int t = 0; t < KT; ++t) {
        // ---- stage B chunk (NT x 1KB) into LDS with async copies ----------
        {
            const char* srcBase = (const char*)wpack;
#pragma unroll
            for (int it = 0; it < NT / 2; ++it) {
                int p = tid + (it << 7);          // [0, NT*64) b128 transfers
                int ntile = p >> 6;
                int q16   = p & 63;
                const char* src = srcBase +
                    (((((size_t)ntile * KT + t) << 9) + ((size_t)q16 << 3)) << 1);
                unsigned dst = bsBase + ((unsigned)p << 4);
                asm volatile("global_load_async_to_lds_b128 %0, %1, off"
                             :: "v"(dst), "v"(src) : "memory");
            }
        }
        // ---- cooperative A-tile fill: 64x32 samples, 16 per thread --------
        {
            int e0 = tid << 4;          // 16 consecutive m, fixed kk
            int kk = e0 >> 6;
            int m0 = e0 & 63;
            int k  = t * 32 + kk;
            int c  = k / K2;
            int k2 = k - c * K2;
            size_t cOff = (size_t)c * HW;
#pragma unroll
            for (int e4 = 0; e4 < 16; ++e4) {
                int m  = m0 + e4;
                int ci = (k2 << 6) + m;
                const float* base = x + ((size_t)sB[m] * C) * HW + cOff;
                float v = sWgt[0][ci] * base[sIdx[0][ci]]
                        + sWgt[1][ci] * base[sIdx[1][ci]]
                        + sWgt[2][ci] * base[sIdx[2][ci]]
                        + sWgt[3][ci] * base[sIdx[3][ci]];
                Alds[(m << 5) + kk] = (_Float16)v;
            }
        }
        asm volatile("s_wait_asynccnt 0x0" ::: "memory");
        __syncthreads();

        // ---- A fragment (contiguous k = 16h + s, as in wpack), reused across NT tiles -
        v16h a;
        {
            const _Float16* Ar = &Alds[(wv * 16 + mA) << 5];
#pragma unroll
            for (int j = 0; j < 8; ++j) {
                a[j]     = Ar[off8 + j];
                a[j + 8] = Ar[off8 + 8 + j];
            }
        }
#pragma unroll
        for (int i = 0; i < NT; ++i) {
            v16h bf = *(const v16h*)&Bs[(i << 9) + (lane << 4)];
            acc[i] = __builtin_amdgcn_wmma_f32_16x16x32_f16(
                false, a, false, bf, (short)0, acc[i], false, false);
        }
        __syncthreads();
    }

    // ---- epilogue: C/D layout is M = r + 8*(lane>>4), N = lane&15 --------
    const int n   = lane & 15;
    const int mhi = (lane >> 4) * 8;
#pragma unroll
    for (int i = 0; i < NT; ++i) {
        int oc = (i << 4) + n;
        float bb = bias[oc];
#pragma unroll
        for (int r = 0; r < 8; ++r) {
            int mloc = wv * 16 + r + mhi;
            int pos  = blockIdx.x * 64 + mloc;
            int b    = sB[mloc];
            int rem  = pos - b * HW;      // h*W + w
            float v  = acc[i][r] + bb;
            if (relu) v = fmaxf(v, 0.0f);
            out[((size_t)b * Co + oc) * HW + rem] = v;
        }
    }
}

// ---------------------------------------------------------------------------
// Host launcher: 8 layers, ping-pong activations in d_ws.
//   d_in: [0]=x, then per layer {wd, bd, wo, bo}.
// ---------------------------------------------------------------------------
extern "C" void kernel_launch(void* const* d_in, const int* in_sizes, int n_in,
                              void* d_out, int out_size, void* d_ws, size_t ws_size,
                              hipStream_t stream) {
    (void)in_sizes; (void)n_in; (void)out_size; (void)ws_size;

    static const int cfg_cin[8]  = {512, 256, 128, 128, 128, 128, 128, 128};
    static const int cfg_cout[8] = {256, 128, 128, 128, 128, 128, 128, 128};
    static const int cfg_ks[8]   = {1, 3, 3, 3, 3, 3, 3, 3};
    static const int cfg_pad[8]  = {0, 1, 1, 1, 1, 1, 1, 1};

    const int B = 2, H = 128, W = 256;
    const int HW = H * W;
    const int M = B * HW;                 // 65536, tiles of 64 -> 1024 blocks

    char* ws = (char*)d_ws;
    float*    bufA = (float*)ws;                              // 16 MB (layer0 out: 256ch)
    float*    bufB = (float*)(ws + ((size_t)16 << 20));       //  8 MB
    float*    offb = (float*)(ws + ((size_t)24 << 20));       //  4.5 MB (max 18ch)
    _Float16* wpk  = (_Float16*)(ws + ((size_t)34 << 20));    //  ~0.6 MB max

    const float* xin = (const float*)d_in[0];

    for (int i = 0; i < 8; ++i) {
        const float* wd = (const float*)d_in[1 + 4 * i + 0];
        const float* bd = (const float*)d_in[1 + 4 * i + 1];
        const float* wo = (const float*)d_in[1 + 4 * i + 2];
        const float* bo = (const float*)d_in[1 + 4 * i + 3];

        const int cin = cfg_cin[i], cout = cfg_cout[i];
        const int ks = cfg_ks[i], pad = cfg_pad[i];
        const int K2 = ks * ks;
        const int K  = cin * K2;

        float* xout = (i == 7) ? (float*)d_out : (((i & 1) == 0) ? bufA : bufB);

        // 1) pack weights -> f16 B-fragment order
        {
            size_t tot = (size_t)cout * K;
            pack_weights_kernel<<<(unsigned)((tot + 255) / 256), 256, 0, stream>>>(
                wd, wpk, cout, K);
        }
        // 2) offset conv
        {
            size_t tot = (size_t)B * (2 * K2) * HW;
            unsigned g = (unsigned)((tot + 255) / 256);
            if (ks == 1)
                offset_conv_kernel<1><<<g, 256, 0, stream>>>(xin, wo, bo, offb,
                                                             B, cin, 2 * K2, H, W, pad);
            else
                offset_conv_kernel<3><<<g, 256, 0, stream>>>(xin, wo, bo, offb,
                                                             B, cin, 2 * K2, H, W, pad);
        }
        // 3) deformable conv as WMMA GEMM (+bias, +ReLU except last)
        {
            int relu = (i < 7) ? 1 : 0;
            unsigned g = (unsigned)(M / 64);
            if (ks == 1)   // Co=256 -> NT=16 tiles, all per wave
                deform_wmma_kernel<1, 1, 16><<<g, 128, 0, stream>>>(
                    xin, offb, wpk, bd, xout, B, cin, cout, H, W, pad, relu);
            else           // Co=128 -> NT=8 tiles, all per wave
                deform_wmma_kernel<9, 3, 8><<<g, 128, 0, stream>>>(
                    xin, offb, wpk, bd, xout, B, cin, cout, H, W, pad, relu);
        }
        xin = xout;
    }
}
